// CRaWl_38001870635467
// MI455X (gfx1250) — matrix operand order repaired and checked by reading, not compiled, not run on hardware
//
#include <hip/hip_runtime.h>
#include <hip/hip_bf16.h>

// ---------------------------------------------------------------------------
// CRaWl walk-GNN on MI455X (gfx1250).
//  * All heavy math runs through v_wmma_f32_16x16x32_f16 (f16 in, f32 acc).
//  * Conv1d(K=9) -> implicit GEMM; one workgroup per walk, input tile in LDS.
//  * Weights pre-packed on device into the exact CDNA5 B-fragment layout so
//    every lane's 16 halves are two contiguous 16B loads.
//  * Training-mode BN => staged kernels with global channel stats (atomics).
// ---------------------------------------------------------------------------

typedef __attribute__((ext_vector_type(16))) _Float16 v16h;
typedef __attribute__((ext_vector_type(8)))  float    v8f;
typedef __attribute__((ext_vector_type(4)))  unsigned int u32x4;
using f16 = _Float16;

union F16x16 { v16h v; u32x4 q[2]; };   // trivially-constructible fragment

static __device__ __forceinline__ v8f wmma_f16(v16h a, v16h b, v8f c) {
  return __builtin_amdgcn_wmma_f32_16x16x32_f16(false, a, false, b, (short)0, c,
                                                false, false);
}

static __device__ __forceinline__ F16x16 load_frag(const f16* p) {
  F16x16 f;
  f.q[0] = *(const u32x4*)p;
  f.q[1] = *(const u32x4*)(p + 16);
  return f;
}

// ---- model constants -------------------------------------------------------
#define NWALK  16384
#define NNODE  16384
#define LWALK  66
#define P1     58          // conv1 output length
#define P2     50          // conv2 output length
#define HDIM   128
#define NGRAPH 128
#define OUTD   10

// ===========================================================================
// small helper kernels
// ===========================================================================
__global__ void k_f32_to_f16(const float* __restrict__ s, f16* __restrict__ d,
                             long n) {
  long i = (long)blockIdx.x * blockDim.x + threadIdx.x;
  if (i < n) d[i] = (f16)s[i];
}

// Pack conv weights w[cout=128][cin][9] into WMMA B fragments.
// Fragment order: frag[(ks*8 + nt)][lane][j], ks = k*(cinp/32)+cb.
__global__ void k_pack_conv_w(const float* __restrict__ wc,
                              f16* __restrict__ frag,
                              int cin, int cinp, int kSteps) {
  int total = kSteps * 8 * 512;
  int idx = blockIdx.x * blockDim.x + threadIdx.x;
  if (idx >= total) return;
  int j    = idx & 15;
  int lane = (idx >> 4) & 31;
  int nt   = (idx >> 9) & 7;
  int ks   = idx >> 12;
  int cb32 = cinp >> 5;
  int k    = ks / cb32;
  int cb   = ks - k * cb32;
  int co   = (nt << 4) + (lane & 15);
  int off  = (lane < 16) ? 0 : 8;
  int kl   = (j < 8) ? (off + j) : (16 + off + (j - 8));
  int ci   = (cb << 5) + kl;
  float v  = (ci < cin) ? wc[(co * cin + ci) * 9 + k] : 0.0f;
  frag[idx] = (f16)v;
}

// Pack dense B[K][N] (row-major) into fragments: frag[(ks*ntiles+nt)][lane][j]
__global__ void k_pack_gemm_b(const float* __restrict__ B,
                              f16* __restrict__ frag, int K, int N) {
  int ntiles = N >> 4;
  int total  = (K >> 5) * ntiles * 512;
  int idx = blockIdx.x * blockDim.x + threadIdx.x;
  if (idx >= total) return;
  int j    = idx & 15;
  int lane = (idx >> 4) & 31;
  int fi   = idx >> 9;
  int nt   = fi % ntiles;
  int ks   = fi / ntiles;
  int n    = (nt << 4) + (lane & 15);
  int off  = (lane < 16) ? 0 : 8;
  int kl   = (j < 8) ? (off + j) : (16 + off + (j - 8));
  int kk   = (ks << 5) + kl;
  frag[idx] = (f16)B[kk * N + n];
}

__global__ void k_bn_finalize(const float* __restrict__ sum,
                              const float* __restrict__ sq,
                              const float* __restrict__ g,
                              const float* __restrict__ b,
                              float* __restrict__ sc, float* __restrict__ sh,
                              float cnt, int C) {
  int c = blockIdx.x * blockDim.x + threadIdx.x;
  if (c >= C) return;
  float m = sum[c] / cnt;
  float v = sq[c] / cnt - m * m;
  float s = g[c] * rsqrtf(v + 1e-5f);
  sc[c] = s;
  sh[c] = b[c] - m * s;
}

__global__ void k_colstats(const float* __restrict__ x, float* __restrict__ sum,
                           float* __restrict__ sq, long total, int N) {
  long i = (long)blockIdx.x * blockDim.x + threadIdx.x;
  long stride = (long)gridDim.x * blockDim.x;
  for (; i < total; i += stride) {
    float v = x[i];
    int c = (int)(i % N);
    atomicAdd(&sum[c], v);
    atomicAdd(&sq[c], v * v);
  }
}

__global__ void k_affine_relu_f16(const float* __restrict__ x,
                                  const float* __restrict__ sc,
                                  const float* __restrict__ sh,
                                  f16* __restrict__ out, long n, int N) {
  long i = (long)blockIdx.x * blockDim.x + threadIdx.x;
  if (i >= n) return;
  int c = (int)(i % N);
  out[i] = (f16)fmaxf(sc[c] * x[i] + sh[c], 0.0f);
}

// ===========================================================================
// Walk conv via implicit-GEMM WMMA.  128 threads (4 waves) per walk.
// mode 0: build tile from gathered node features + walk_x (both f16)
// mode 1: build tile from conv1 output with BN-affine + ReLU applied
// M tile = 64 positions (padded), N = 128 channels, K step = 32 channels.
// ===========================================================================
__global__ void __launch_bounds__(128)
k_walk_conv(const int* __restrict__ walk_nodes,
            const f16* __restrict__ nodef, int din,
            const f16* __restrict__ wx16,
            const f16* __restrict__ c1in,
            const float* __restrict__ inScale,
            const float* __restrict__ inShift,
            const f16* __restrict__ wfrag,
            int cinp, int kSteps,
            f16* __restrict__ outp, int P,
            float* __restrict__ gSum, float* __restrict__ gSq, int mode) {
  extern __shared__ char smem_raw[];
  f16*   Xs   = (f16*)smem_raw;
  float* sSum = (float*)(smem_raw + 72 * cinp * 2);
  float* sSq  = sSum + 128;

  const int w   = blockIdx.x;
  const int tid = threadIdx.x;

  // ---- build input tile [72 rows x cinp ch] in LDS (rows >= valid are 0) ----
  if (mode == 0) {
    const int cin = din + 24;
    for (int idx = tid; idx < LWALK * cinp; idx += 128) {
      int l = idx / cinp, c = idx - l * cinp;
      f16 v = (f16)0.0f;
      if (c < din) {
        int node = walk_nodes[w * LWALK + l];
        v = nodef[(size_t)node * din + c];
      } else if (c < cin) {
        v = wx16[((size_t)w * 24 + (c - din)) * LWALK + l];
      }
      Xs[idx] = v;
    }
  } else {
    for (int idx = tid; idx < LWALK * cinp; idx += 128) {
      int l = idx / cinp, c = idx - l * cinp;
      f16 v = (f16)0.0f;
      if (l < P1) {
        float xv = inScale[c] * (float)c1in[((size_t)w * P1 + l) * HDIM + c] +
                   inShift[c];
        v = (f16)fmaxf(xv, 0.0f);
      }
      Xs[idx] = v;
    }
  }
  for (int idx = LWALK * cinp + tid; idx < 72 * cinp; idx += 128)
    Xs[idx] = (f16)0.0f;
  sSum[tid] = 0.0f;
  sSq[tid]  = 0.0f;
  __syncthreads();

  // ---- WMMA main loop ----
  const int wave  = tid >> 5;
  const int lane  = tid & 31;
  const int mbase = wave << 4;                 // 4 waves cover 64 positions
  const int r0    = mbase + (lane & 15);
  const int off   = (lane < 16) ? 0 : 8;
  const int cb32  = cinp >> 5;

  v8f acc[8] = {};
  int k = 0, cb = 0;
  for (int ks = 0; ks < kSteps; ++ks) {
    const f16* ap = &Xs[(r0 + k) * cinp + (cb << 5) + off];
    F16x16 A = load_frag(ap);
    const f16* bp = &wfrag[((ks << 3) * 32 + lane) << 4];
#pragma unroll
    for (int nt = 0; nt < 8; ++nt) {
      F16x16 B;
      B.q[0] = *(const u32x4*)(bp + (nt << 9));
      B.q[1] = *(const u32x4*)(bp + (nt << 9) + 8);
      acc[nt] = wmma_f16(A.v, B.v, acc[nt]);
    }
    if (++cb == cb32) { cb = 0; ++k; }
  }

  // ---- store valid rows (f16) + channel stats for BN ----
  const int mofs = mbase + ((lane >> 4) << 3);
#pragma unroll
  for (int nt = 0; nt < 8; ++nt) {
    int n = (nt << 4) + (lane & 15);
#pragma unroll
    for (int i = 0; i < 8; ++i) {
      int m = mofs + i;
      if (m < P) {
        float v = acc[nt][i];
        outp[((size_t)w * P + m) * HDIM + n] = (f16)v;
        atomicAdd(&sSum[n], v);
        atomicAdd(&sSq[n], v * v);
      }
    }
  }
  __syncthreads();
  atomicAdd(&gSum[tid], sSum[tid]);
  atomicAdd(&gSq[tid], sSq[tid]);
}

// ===========================================================================
// Generic WMMA GEMM: C[M,N] = A[M,K](f16,row-major) * Bfrag + optional resid.
// One wave per block; block computes a 16 x 64 tile.
// ===========================================================================
__global__ void __launch_bounds__(32)
k_gemm_wmma(const f16* __restrict__ A, const f16* __restrict__ Bfrag,
            int N, int K, const float* __restrict__ resid,
            float* __restrict__ Cout) {
  const int mt     = blockIdx.x;
  const int ntile0 = blockIdx.y << 2;
  const int lane   = threadIdx.x;
  const int ntiles = N >> 4;
  const int r      = (mt << 4) + (lane & 15);
  const int off    = (lane < 16) ? 0 : 8;

  v8f acc[4] = {};
  const int kSteps = K >> 5;
  for (int ks = 0; ks < kSteps; ++ks) {
    F16x16 Af = load_frag(&A[(size_t)r * K + (ks << 5) + off]);
#pragma unroll
    for (int nb = 0; nb < 4; ++nb) {
      const f16* bp = &Bfrag[(size_t)((ks * ntiles + ntile0 + nb) * 32 + lane) << 4];
      F16x16 Bf;
      Bf.q[0] = *(const u32x4*)bp;
      Bf.q[1] = *(const u32x4*)(bp + 8);
      acc[nb] = wmma_f16(Af.v, Bf.v, acc[nb]);
    }
  }
  const int mofs = (mt << 4) + ((lane >> 4) << 3);
#pragma unroll
  for (int nb = 0; nb < 4; ++nb) {
    int n = ((ntile0 + nb) << 4) + (lane & 15);
#pragma unroll
    for (int i = 0; i < 8; ++i) {
      int m = mofs + i;
      float v = acc[nb][i];
      if (resid) v += resid[(size_t)m * N + n];
      Cout[(size_t)m * N + n] = v;
    }
  }
}

// ===========================================================================
// scatter-mean pieces
// ===========================================================================
__global__ void __launch_bounds__(128)
k_scatter(const f16* __restrict__ c2, const int* __restrict__ walk_nodes,
          const float* __restrict__ sc, const float* __restrict__ sh,
          float* __restrict__ nodeSum, float* __restrict__ nodeCnt) {
  const int w = blockIdx.x, c = threadIdx.x;
  float s = sc[c], t = sh[c];
  for (int p = 0; p < P2; ++p) {
    int node = walk_nodes[w * LWALK + 8 + p];
    float v = fmaxf(s * (float)c2[((size_t)w * P2 + p) * HDIM + c] + t, 0.0f);
    atomicAdd(&nodeSum[(size_t)node * HDIM + c], v);
    if (c == 0) atomicAdd(&nodeCnt[node], 1.0f);
  }
}

__global__ void k_hm_finalize(const float* __restrict__ nodeSum,
                              const float* __restrict__ nodeCnt,
                              f16* __restrict__ hm16, long n) {
  long i = (long)blockIdx.x * blockDim.x + threadIdx.x;
  if (i >= n) return;
  float cnt = fmaxf(nodeCnt[i >> 7], 1.0f);
  hm16[i] = (f16)(nodeSum[i] / cnt);
}

__global__ void __launch_bounds__(128)
k_graph_pool(const float* __restrict__ h, const float* __restrict__ sc,
             const float* __restrict__ sh, const int* __restrict__ batch,
             float* __restrict__ gS, float* __restrict__ gC) {
  const int node = blockIdx.x, c = threadIdx.x;
  float v = fmaxf(sc[c] * h[(size_t)node * HDIM + c] + sh[c], 0.0f);
  int g = batch[node];
  atomicAdd(&gS[(size_t)g * HDIM + c], v);
  if (c == 0) atomicAdd(&gC[g], 1.0f);
}

__global__ void __launch_bounds__(128)
k_head(const float* __restrict__ gS, const float* __restrict__ gC,
       const float* __restrict__ w1, const float* __restrict__ b1,
       const float* __restrict__ w2, const float* __restrict__ b2,
       float* __restrict__ out) {
  __shared__ float xg[HDIM];
  __shared__ float y1[HDIM];
  const int g = blockIdx.x, t = threadIdx.x;
  float cnt = fmaxf(gC[g], 1.0f);
  xg[t] = gS[g * HDIM + t] / cnt;
  __syncthreads();
  float s = b1[t];
  for (int c = 0; c < HDIM; ++c) s += xg[c] * w1[c * HDIM + t];
  y1[t] = fmaxf(s, 0.0f);
  __syncthreads();
  if (t < OUTD) {
    float o = b2[t];
    for (int c = 0; c < HDIM; ++c) o += y1[c] * w2[c * OUTD + t];
    out[g * OUTD + t] = o;
  }
}

// ===========================================================================
// host orchestration
// ===========================================================================
extern "C" void kernel_launch(void* const* d_in, const int* in_sizes, int n_in,
                              void* d_out, int out_size, void* d_ws,
                              size_t ws_size, hipStream_t stream) {
  (void)n_in; (void)out_size; (void)ws_size;
  const float* x      = (const float*)d_in[0];
  const float* walk_x = (const float*)d_in[1];

  const float *b1[3], *b2l[3], *bo[3], *g1[3], *g2[3], *go[3];
  const float *wc1[3], *wc2[3], *wo1[3], *wo2[3];
  const float *node_b, *node_g, *out_b1, *out_b2, *out_w1, *out_w2, *rescale;
  const int *walk_nodes, *batch;

  if (in_sizes[2] == 128) {
    // JAX sorted-key pytree flatten: layers first, each dict key-sorted.
    int p = 2;
    for (int i = 0; i < 3; ++i) {
      b1[i]  = (const float*)d_in[p++]; b2l[i] = (const float*)d_in[p++];
      bo[i]  = (const float*)d_in[p++]; g1[i]  = (const float*)d_in[p++];
      g2[i]  = (const float*)d_in[p++]; go[i]  = (const float*)d_in[p++];
      wc1[i] = (const float*)d_in[p++]; wc2[i] = (const float*)d_in[p++];
      wo1[i] = (const float*)d_in[p++]; wo2[i] = (const float*)d_in[p++];
    }
    node_b = (const float*)d_in[p++]; node_g = (const float*)d_in[p++];
    out_b1 = (const float*)d_in[p++]; out_b2 = (const float*)d_in[p++];
    out_w1 = (const float*)d_in[p++]; out_w2 = (const float*)d_in[p++];
    rescale = (const float*)d_in[p++];
    walk_nodes = (const int*)d_in[p++]; batch = (const int*)d_in[p++];
  } else {
    // dict insertion-order flatten: 'rescale' first.
    int p = 2;
    rescale = (const float*)d_in[p++];
    for (int i = 0; i < 3; ++i) {
      wc1[i] = (const float*)d_in[p++]; g1[i]  = (const float*)d_in[p++];
      b1[i]  = (const float*)d_in[p++]; wc2[i] = (const float*)d_in[p++];
      g2[i]  = (const float*)d_in[p++]; b2l[i] = (const float*)d_in[p++];
      wo1[i] = (const float*)d_in[p++]; go[i]  = (const float*)d_in[p++];
      bo[i]  = (const float*)d_in[p++]; wo2[i] = (const float*)d_in[p++];
    }
    node_g = (const float*)d_in[p++]; node_b = (const float*)d_in[p++];
    out_w1 = (const float*)d_in[p++]; out_b1 = (const float*)d_in[p++];
    out_w2 = (const float*)d_in[p++]; out_b2 = (const float*)d_in[p++];
    walk_nodes = (const int*)d_in[p++]; batch = (const int*)d_in[p++];
  }

  // ---- carve workspace ----
  size_t off = 0;
  char* base = (char*)d_ws;
  auto carve = [&](size_t bytes) -> char* {
    char* p = base + off;
    off += (bytes + 255) & ~(size_t)255;
    return p;
  };
  f16*   wx16    = (f16*)carve((size_t)NWALK * 24 * LWALK * 2);
  f16*   h16     = (f16*)carve((size_t)NNODE * HDIM * 2);
  float* hA      = (float*)carve((size_t)NNODE * HDIM * 4);
  float* hB      = (float*)carve((size_t)NNODE * HDIM * 4);
  float* hr      = (float*)carve((size_t)NNODE * HDIM * 4);
  f16*   c1      = (f16*)carve((size_t)NWALK * P1 * HDIM * 2);
  f16*   c2      = (f16*)carve((size_t)NWALK * P2 * HDIM * 2);
  float* nodeSum = (float*)carve((size_t)NNODE * HDIM * 4);
  float* nodeCnt = (float*)carve((size_t)NNODE * 4);
  f16*   hm16    = (f16*)carve((size_t)NNODE * HDIM * 2);
  float* ybuf    = (float*)carve((size_t)NNODE * 256 * 4);
  f16*   z16     = (f16*)carve((size_t)NNODE * 256 * 2);
  // stats block (zeroed once per layer): sum1,sq1,sum2,sq2 (128 ea), sumo,sqo (256 ea)
  float* statsZ  = (float*)carve(1024 * 4);
  float* sum1 = statsZ;       float* sq1 = statsZ + 128;
  float* sum2 = statsZ + 256; float* sq2 = statsZ + 384;
  float* sumo = statsZ + 512; float* sqo = statsZ + 768;
  float* sc1 = (float*)carve(128 * 4); float* sh1 = (float*)carve(128 * 4);
  float* sc2 = (float*)carve(128 * 4); float* sh2 = (float*)carve(128 * 4);
  float* sco = (float*)carve(256 * 4); float* sho = (float*)carve(256 * 4);
  float* scn = (float*)carve(128 * 4); float* shn = (float*)carve(128 * 4);
  float* gPoolS = (float*)carve((size_t)NGRAPH * HDIM * 4);
  float* gPoolC = (float*)carve((size_t)NGRAPH * 4);
  f16* rescaleF = (f16*)carve(1 * 8 * 512 * 2);
  f16* wc1F[3], *wc2F[3], *wo1F[3], *wo2F[3];
  for (int i = 0; i < 3; ++i) {
    int ks1 = (i == 0) ? 18 : 45;               // 9*cinp/32
    wc1F[i] = (f16*)carve((size_t)ks1 * 8 * 512 * 2);
    wc2F[i] = (f16*)carve((size_t)36 * 8 * 512 * 2);
    wo1F[i] = (f16*)carve((size_t)4 * 16 * 512 * 2);   // K=128,N=256
    wo2F[i] = (f16*)carve((size_t)8 * 8 * 512 * 2);    // K=256,N=128
  }

  const int T = 256;
  auto gridOf = [](long n, int t) { return (unsigned)((n + t - 1) / t); };

  // ---- one-time prep (deterministic each call) ----
  {
    long n = (long)NWALK * 24 * LWALK;
    k_f32_to_f16<<<gridOf(n, T), T, 0, stream>>>(walk_x, wx16, n);
  }
  k_pack_gemm_b<<<gridOf(1 * 8 * 512, T), T, 0, stream>>>(rescale, rescaleF, 32, 128);
  for (int i = 0; i < 3; ++i) {
    int cin1 = (i == 0) ? 56 : 152, cinp1 = (i == 0) ? 64 : 160;
    int ks1 = 9 * cinp1 / 32;
    k_pack_conv_w<<<gridOf((long)ks1 * 8 * 512, T), T, 0, stream>>>(wc1[i], wc1F[i], cin1, cinp1, ks1);
    k_pack_conv_w<<<gridOf((long)36 * 8 * 512, T), T, 0, stream>>>(wc2[i], wc2F[i], 128, 128, 36);
    k_pack_gemm_b<<<gridOf((long)4 * 16 * 512, T), T, 0, stream>>>(wo1[i], wo1F[i], 128, 256);
    k_pack_gemm_b<<<gridOf((long)8 * 8 * 512, T), T, 0, stream>>>(wo2[i], wo2F[i], 256, 128);
  }

  // h_r = x @ rescale  (convert x -> f16 first; h16 holds current gather feats)
  {
    long n = (long)NNODE * 32;
    k_f32_to_f16<<<gridOf(n, T), T, 0, stream>>>(x, h16, n);
    k_gemm_wmma<<<dim3(NNODE / 16, 128 / 64), 32, 0, stream>>>(h16, rescaleF, 128, 32, nullptr, hr);
  }

  const float* gatherSrc = x;   // f32 source of current node features
  float* hCur = nullptr;
  int din = 32;

  for (int i = 0; i < 3; ++i) {
    // node features -> f16 (row stride = din)
    long n = (long)NNODE * din;
    k_f32_to_f16<<<gridOf(n, T), T, 0, stream>>>(gatherSrc, h16, n);
    hipMemsetAsync(statsZ, 0, 1024 * 4, stream);

    int cinp1 = (din == 32) ? 64 : 160;
    int ks1 = 9 * cinp1 / 32;
    size_t smem1 = (size_t)72 * cinp1 * 2 + 256 * 4;
    k_walk_conv<<<NWALK, 128, smem1, stream>>>(walk_nodes, h16, din, wx16,
                                               nullptr, nullptr, nullptr,
                                               wc1F[i], cinp1, ks1, c1, P1,
                                               sum1, sq1, 0);
    k_bn_finalize<<<1, 128, 0, stream>>>(sum1, sq1, g1[i], b1[i], sc1, sh1,
                                         (float)NWALK * P1, 128);
    size_t smem2 = (size_t)72 * 128 * 2 + 256 * 4;
    k_walk_conv<<<NWALK, 128, smem2, stream>>>(walk_nodes, nullptr, 128, nullptr,
                                               c1, sc1, sh1,
                                               wc2F[i], 128, 36, c2, P2,
                                               sum2, sq2, 1);
    k_bn_finalize<<<1, 128, 0, stream>>>(sum2, sq2, g2[i], b2l[i], sc2, sh2,
                                         (float)NWALK * P2, 128);

    hipMemsetAsync(nodeSum, 0, (size_t)NNODE * HDIM * 4, stream);
    hipMemsetAsync(nodeCnt, 0, (size_t)NNODE * 4, stream);
    k_scatter<<<NWALK, 128, 0, stream>>>(c2, walk_nodes, sc2, sh2, nodeSum, nodeCnt);
    {
      long m = (long)NNODE * HDIM;
      k_hm_finalize<<<gridOf(m, T), T, 0, stream>>>(nodeSum, nodeCnt, hm16, m);
    }

    // y = hm @ wo1  [16384,256]
    k_gemm_wmma<<<dim3(NNODE / 16, 256 / 64), 32, 0, stream>>>(hm16, wo1F[i], 256, 128, nullptr, ybuf);
    k_colstats<<<2048, T, 0, stream>>>(ybuf, sumo, sqo, (long)NNODE * 256, 256);
    k_bn_finalize<<<2, 128, 0, stream>>>(sumo, sqo, go[i], bo[i], sco, sho,
                                         (float)NNODE, 256);
    {
      long m = (long)NNODE * 256;
      k_affine_relu_f16<<<gridOf(m, T), T, 0, stream>>>(ybuf, sco, sho, z16, m, 256);
    }
    // h_new = z @ wo2 + h_r
    float* hNext = (i & 1) ? hB : hA;
    const float* res = (i == 0) ? hr : hCur;
    k_gemm_wmma<<<dim3(NNODE / 16, 128 / 64), 32, 0, stream>>>(z16, wo2F[i], 128, 256, res, hNext);

    gatherSrc = hNext;
    hCur = hNext;
    din = 128;
  }

  // final node BN + relu, graph mean-pool, head MLP
  hipMemsetAsync(statsZ, 0, 256 * 4, stream);                  // sum1, sq1
  k_colstats<<<2048, T, 0, stream>>>(hCur, sum1, sq1, (long)NNODE * HDIM, HDIM);
  k_bn_finalize<<<1, 128, 0, stream>>>(sum1, sq1, node_g, node_b, scn, shn,
                                       (float)NNODE, HDIM);
  hipMemsetAsync(gPoolS, 0, (size_t)NGRAPH * HDIM * 4, stream);
  hipMemsetAsync(gPoolC, 0, (size_t)NGRAPH * 4, stream);
  k_graph_pool<<<NNODE, 128, 0, stream>>>(hCur, scn, shn, batch, gPoolS, gPoolC);
  k_head<<<NGRAPH, 128, 0, stream>>>(gPoolS, gPoolC, out_w1, out_b1, out_w2,
                                     out_b2, (float*)d_out);
}
